// Attention_9560597201104
// MI455X (gfx1250) — compile-verified, matrix operand order checked by reading, not run on hardware
//
#include <hip/hip_runtime.h>

typedef __attribute__((ext_vector_type(16))) _Float16 v16h;
typedef __attribute__((ext_vector_type(8)))  _Float16 v8h;
typedef __attribute__((ext_vector_type(8)))  float    v8f;
typedef __attribute__((ext_vector_type(4)))  float    v4f;

#define B_    128
#define N_    197
#define NP_   208     // padded token count (13 * 16)
#define C_    768
#define H_    12
#define HD_   64
#define C3_   2304
#define SROW  224     // padded score row (7 * 32) for WMMA K-loop
#define QKV_NCT 8     // column tiles per wave (QKV gemm)
#define PRJ_NCT 8     // column tiles per wave (proj gemm)

static __device__ __forceinline__ v8f wmma16x16x32(v16h a, v16h b, v8f c) {
  return __builtin_amdgcn_wmma_f32_16x16x32_f16(false, a, false, b,
                                                (short)0, c, false, false);
}

// 16-bit A/B fragment of v_wmma_*_16x16x32 (ISA 7.12.2): per lane the 16
// elements are two contiguous 8-element runs: k = hi*8 + [0,8) and
// k = hi*8 + 16 + [0,8).  Load as two b128s.
static __device__ __forceinline__ v16h frag_f16(const _Float16* p, int hi) {
  const v8h* q = (const v8h*)(p + hi * 8);
  const v8h lo = q[0];   // k = hi*8 + 0..7
  const v8h hh = q[2];   // k = hi*8 + 16..23
  return __builtin_shufflevector(lo, hh, 0, 1, 2, 3, 4, 5, 6, 7,
                                 8, 9, 10, 11, 12, 13, 14, 15);
}

// Same fragment sourced from f32 memory (four b128 loads + cvt).
static __device__ __forceinline__ v16h frag_f32(const float* p, int hi) {
  const v4f* q = (const v4f*)(p + hi * 8);
  const v4f a0 = q[0], a1 = q[1];   // k = hi*8 + 0..7
  const v4f b0 = q[4], b1 = q[5];   // k = hi*8 + 16..23
  v16h r;
#pragma unroll
  for (int e = 0; e < 4; ++e) {
    r[e]      = (_Float16)a0[e];
    r[e + 4]  = (_Float16)a1[e];
    r[e + 8]  = (_Float16)b0[e];
    r[e + 12] = (_Float16)b1[e];
  }
  return r;
}

// ---------------------------------------------------------------------------
// Kernel 1: weight f32->f16 conversion + relative-position-bias gather
// ---------------------------------------------------------------------------
__global__ void prep_kernel(const float* __restrict__ wqkv,
                            const float* __restrict__ wproj,
                            const float* __restrict__ table,
                            const int*   __restrict__ relidx,
                            _Float16* __restrict__ wq16,
                            _Float16* __restrict__ wp16,
                            float*    __restrict__ rpb) {
  const int i  = blockIdx.x * blockDim.x + threadIdx.x;
  const int nq = C3_ * C_;
  const int np = C_ * C_;
  const int nr = H_ * N_ * N_;
  if (i < nq) {
    wq16[i] = (_Float16)wqkv[i];
  } else if (i < nq + np) {
    wp16[i - nq] = (_Float16)wproj[i - nq];
  } else if (i < nq + np + nr) {
    const int t  = i - nq - np;
    const int h  = t / (N_ * N_);
    const int ij = t % (N_ * N_);
    rpb[t] = table[relidx[ij] * H_ + h];   // layout [H][N][N]
  }
}

// ---------------------------------------------------------------------------
// Kernel 2: QKV projection (x @ Wqkv^T + bitfit bias), q scaled, f16 out
// grid: (18, 13, 128) x (32)  -> each wave: 16 rows x (8*16) cols
// ---------------------------------------------------------------------------
__global__ void __launch_bounds__(32) qkv_kernel(
    const float* __restrict__ x, const int* __restrict__ bidx,
    const _Float16* __restrict__ wq16,
    const float* __restrict__ qb, const float* __restrict__ kb,
    const float* __restrict__ vb,
    _Float16* __restrict__ Q, _Float16* __restrict__ K,
    _Float16* __restrict__ V) {
  const int lane = threadIdx.x;
  const int ln15 = lane & 15;             // A-row / B-col / D-col within tile
  const int hi   = lane >> 4;
  const int ctg  = blockIdx.x * QKV_NCT;  // first of 144 column tiles
  const int rt   = blockIdx.y;            // row tile 0..12
  const int b    = blockIdx.z;

  v8f acc[QKV_NCT];
#pragma unroll
  for (int t = 0; t < QKV_NCT; ++t) acc[t] = {};

  // Rows >= N_ are clamped to row 0: garbage in A-row m only affects D-row m,
  // and all token >= N_ rows are forced to zero in the epilogue.
  const int tok = rt * 16 + ln15;
  const float* xr = x + ((size_t)b * N_ + (tok < N_ ? tok : 0)) * C_;

  for (int kk = 0; kk < C_; kk += 32) {
    if (kk + 32 < C_) __builtin_prefetch(xr + kk + 32, 0, 0);
    const v16h a = frag_f32(xr + kk, hi);
#pragma unroll
    for (int t = 0; t < QKV_NCT; ++t) {
      const _Float16* wr = wq16 + (size_t)((ctg + t) * 16 + ln15) * C_ + kk;
      if (kk + 32 < C_) __builtin_prefetch(wr + 32, 0, 0);
      acc[t] = wmma16x16x32(a, frag_f16(wr, hi), acc[t]);
    }
  }

  const int set = bidx[b];
#pragma unroll
  for (int t = 0; t < QKV_NCT; ++t) {
    const int colg = (ctg + t) * 16 + ln15;     // 0..2303
    const int sec  = colg / C_;                 // 0=q 1=k 2=v
    const int cw   = colg % C_;
    const int h    = cw / HD_;
    const int d    = cw % HD_;
    const float bias = (sec == 0 ? qb : (sec == 1 ? kb : vb))[set * C_ + cw];
    _Float16* dst = (sec == 0 ? Q : (sec == 1 ? K : V));
#pragma unroll
    for (int r = 0; r < 8; ++r) {
      const int tk = rt * 16 + r + 8 * hi;
      float v2 = acc[t][r] + bias;
      if (sec == 0) v2 *= 0.125f;               // 1/sqrt(64)
      if (tk >= N_) v2 = 0.f;                   // clean zero padding
      dst[(((size_t)b * H_ + h) * NP_ + tk) * HD_ + d] = (_Float16)v2;
    }
  }
}

// ---------------------------------------------------------------------------
// Kernel 3: attention, one workgroup per (b,h); Q,K,V^T,P in LDS
// grid: (12, 128) x (256)
// ---------------------------------------------------------------------------
__global__ void __launch_bounds__(256) attn_kernel(
    const _Float16* __restrict__ Q, const _Float16* __restrict__ K,
    const _Float16* __restrict__ V, const float* __restrict__ rpb,
    _Float16* __restrict__ ctx) {
  extern __shared__ char smem[];
  _Float16* Qs   = (_Float16*)smem;            // [208][64]
  _Float16* Ks   = Qs + NP_ * HD_;             // [208][64]
  _Float16* Vt   = Ks + NP_ * HD_;             // [64][224]  (transposed V)
  float*    Ssum = (float*)(Vt + HD_ * SROW);  // [208]
  _Float16* P    = (_Float16*)(Ssum + NP_);    // [208][224] scores -> probs

  const int h = blockIdx.x, b = blockIdx.y;
  const int tid = threadIdx.x;
  const size_t base = ((size_t)b * H_ + h) * NP_ * HD_;

  // Stage Q/K with 16B vector copies.
  {
    const uint4* gq = (const uint4*)(Q + base);
    const uint4* gk = (const uint4*)(K + base);
    uint4* sq = (uint4*)Qs;
    uint4* sk = (uint4*)Ks;
    for (int i = tid; i < NP_ * HD_ / 8; i += 256) {
      sq[i] = gq[i];
      sk[i] = gk[i];
    }
    // Stage V transposed: read 8 f16 along d, scatter into Vt[d][tok].
    const v8h* gv = (const v8h*)(V + base);
    for (int i = tid; i < NP_ * HD_ / 8; i += 256) {
      const int tok = i >> 3;
      const int d0  = (i & 7) * 8;
      const v8h v = gv[i];
#pragma unroll
      for (int e = 0; e < 8; ++e) Vt[(d0 + e) * SROW + tok] = v[e];
    }
    // Zero Vt pad columns 208..223 so P(=0)*Vt(pad) stays 0 (never NaN).
    for (int i = tid; i < HD_ * (SROW - NP_); i += 256)
      Vt[(i >> 4) * SROW + NP_ + (i & 15)] = (_Float16)0.f;
  }
  __syncthreads();

  const int wave = tid >> 5, lane = tid & 31;
  const int ln15 = lane & 15, hi = lane >> 4;

  // S = Q K^T + rpb  (13x13 tiles, wave-strided)
  for (int t = wave; t < 13 * 13; t += 8) {
    const int it = t / 13, jt = t % 13;
    v8f acc = {};
#pragma unroll
    for (int kk = 0; kk < HD_; kk += 32) {
      const v16h a  = frag_f16(Qs + (it * 16 + ln15) * HD_ + kk, hi);
      const v16h bb = frag_f16(Ks + (jt * 16 + ln15) * HD_ + kk, hi);
      acc = wmma16x16x32(a, bb, acc);
    }
#pragma unroll
    for (int r = 0; r < 8; ++r) {
      const int i = it * 16 + r + 8 * hi;
      const int j = jt * 16 + ln15;
      float v2 = acc[r];
      if (i < N_ && j < N_) v2 += rpb[(h * N_ + i) * N_ + j];
      P[i * SROW + j] = (j < N_) ? (_Float16)v2 : (_Float16)(-65504.f);
    }
  }
  __syncthreads();

  // row softmax (unnormalized exp back into P; 1/sum folded into epilogue)
  if (tid < NP_) {
    _Float16* prow = P + tid * SROW;
    if (tid < N_) {
      float mx = -1e30f;
      for (int j = 0; j < NP_; ++j) mx = fmaxf(mx, (float)prow[j]);
      float s = 0.f;
      for (int j = 0; j < NP_; ++j) {
        const float e2 = __expf((float)prow[j] - mx);
        s += e2;
        prow[j] = (_Float16)e2;
      }
      for (int j = NP_; j < SROW; ++j) prow[j] = (_Float16)0.f;
      Ssum[tid] = s;
    } else {
      for (int j = 0; j < SROW; ++j) prow[j] = (_Float16)0.f;
      Ssum[tid] = 1.f;
    }
  }
  __syncthreads();

  // O = P V  (13 i-tiles x 4 d-tiles, wave-strided), scale by 1/rowsum
  for (int t = wave; t < 13 * 4; t += 8) {
    const int it = t >> 2, dt = t & 3;
    v8f acc = {};
#pragma unroll
    for (int j0 = 0; j0 < SROW; j0 += 32) {
      const v16h a  = frag_f16(P + (it * 16 + ln15) * SROW + j0, hi);
      const v16h bb = frag_f16(Vt + (dt * 16 + ln15) * SROW + j0, hi);
      acc = wmma16x16x32(a, bb, acc);
    }
#pragma unroll
    for (int r = 0; r < 8; ++r) {
      const int i = it * 16 + r + 8 * hi;
      if (i < N_) {
        const float o = acc[r] / Ssum[i];
        ctx[((size_t)b * N_ + i) * C_ + h * HD_ + dt * 16 + ln15] = (_Float16)o;
      }
    }
  }
}

// ---------------------------------------------------------------------------
// Kernel 4: output projection (ctx @ Wproj^T + b_proj[b_idx]) -> f32
// grid: (6, 13, 128) x (32)
// ---------------------------------------------------------------------------
__global__ void __launch_bounds__(32) proj_kernel(
    const _Float16* __restrict__ ctx, const _Float16* __restrict__ wp16,
    const float* __restrict__ bproj, const int* __restrict__ bidx,
    float* __restrict__ out) {
  const int lane = threadIdx.x;
  const int ln15 = lane & 15, hi = lane >> 4;
  const int ctg  = blockIdx.x * PRJ_NCT;   // first of 48 column tiles
  const int rt   = blockIdx.y;
  const int b    = blockIdx.z;

  const int tok = rt * 16 + ln15;
  const _Float16* xr = ctx + ((size_t)b * N_ + (tok < N_ ? tok : 0)) * C_;

  v8f acc[PRJ_NCT];
#pragma unroll
  for (int t = 0; t < PRJ_NCT; ++t) acc[t] = {};

  for (int kk = 0; kk < C_; kk += 32) {
    if (kk + 32 < C_) __builtin_prefetch(xr + kk + 32, 0, 0);
    const v16h a = frag_f16(xr + kk, hi);
#pragma unroll
    for (int t = 0; t < PRJ_NCT; ++t) {
      const _Float16* wr = wp16 + (size_t)((ctg + t) * 16 + ln15) * C_ + kk;
      if (kk + 32 < C_) __builtin_prefetch(wr + 32, 0, 0);
      acc[t] = wmma16x16x32(a, frag_f16(wr, hi), acc[t]);
    }
  }

  const int set = bidx[b];
#pragma unroll
  for (int t = 0; t < PRJ_NCT; ++t) {
    const int colg = (ctg + t) * 16 + ln15;
    const float bias = bproj[set * C_ + colg];
#pragma unroll
    for (int r = 0; r < 8; ++r) {
      const int tk = rt * 16 + r + 8 * hi;
      if (tk < N_)
        out[((size_t)b * N_ + tk) * C_ + colg] = acc[t][r] + bias;
    }
  }
}

// ---------------------------------------------------------------------------
extern "C" void kernel_launch(void* const* d_in, const int* in_sizes, int n_in,
                              void* d_out, int out_size, void* d_ws, size_t ws_size,
                              hipStream_t stream) {
  (void)in_sizes; (void)n_in; (void)out_size; (void)ws_size;
  const float* x      = (const float*)d_in[0];
  const int*   bidx   = (const int*)  d_in[1];
  const float* wqkv   = (const float*)d_in[2];
  const float* qb     = (const float*)d_in[3];
  const float* kb     = (const float*)d_in[4];
  const float* vb     = (const float*)d_in[5];
  const float* table  = (const float*)d_in[6];
  const int*   relidx = (const int*)  d_in[7];
  const float* wproj  = (const float*)d_in[8];
  const float* bproj  = (const float*)d_in[9];
  float* out = (float*)d_out;

  char* ws = (char*)d_ws;
  size_t off = 0;
  auto carve = [&](size_t bytes) -> char* {
    char* p = ws + off;
    off += (bytes + 255) & ~(size_t)255;
    return p;
  };
  _Float16* wq16 = (_Float16*)carve((size_t)C3_ * C_ * 2);
  _Float16* wp16 = (_Float16*)carve((size_t)C_ * C_ * 2);
  float*    rpb  = (float*)   carve((size_t)H_ * N_ * N_ * 4);
  _Float16* Qb   = (_Float16*)carve((size_t)B_ * H_ * NP_ * HD_ * 2);
  _Float16* Kb   = (_Float16*)carve((size_t)B_ * H_ * NP_ * HD_ * 2);
  _Float16* Vb   = (_Float16*)carve((size_t)B_ * H_ * NP_ * HD_ * 2);
  _Float16* ctx  = (_Float16*)carve((size_t)B_ * N_ * C_ * 2);

  {
    const int n = C3_ * C_ + C_ * C_ + H_ * N_ * N_;
    prep_kernel<<<(n + 255) / 256, 256, 0, stream>>>(wqkv, wproj, table, relidx,
                                                     wq16, wp16, rpb);
  }
  qkv_kernel<<<dim3(C3_ / 16 / QKV_NCT, 13, B_), 32, 0, stream>>>(
      x, bidx, wq16, qb, kb, vb, Qb, Kb, Vb);

  const size_t smem = (size_t)2 * NP_ * HD_ * 2 +       // Qs, Ks
                      (size_t)HD_ * SROW * 2 +          // Vt
                      (size_t)NP_ * 4 +                 // Ssum
                      (size_t)NP_ * SROW * 2;           // P   (~176 KB < 320 KB)
  (void)hipFuncSetAttribute((const void*)attn_kernel,
                            hipFuncAttributeMaxDynamicSharedMemorySize,
                            (int)smem);
  attn_kernel<<<dim3(H_, B_), 256, smem, stream>>>(Qb, Kb, Vb, rpb, ctx);

  proj_kernel<<<dim3(C_ / 16 / PRJ_NCT, 13, B_), 32, 0, stream>>>(
      ctx, wp16, bproj, bidx, out);
}
